// MoE_6889127543053
// MI455X (gfx1250) — matrix operand order repaired and checked by reading, not compiled, not run on hardware
//
#include <hip/hip_runtime.h>
#include <hip/hip_bf16.h>
#include <math.h>

// ---------------- problem constants ----------------
constexpr int kD = 1024;   // model dim
constexpr int kH = 2048;   // expert hidden dim
constexpr int kE = 8;      // experts
constexpr int kT = 8192;   // tokens (B*S)
constexpr int kMT = 32;    // tokens per FFN tile
constexpr int kHC = 64;    // hidden-chunk per phase
constexpr int kKS = 128;   // k staging depth for fc1

// ---------------- LDS layout (dynamic shared) ----------------
constexpr int XS_STR  = kD + 8;     // 1032 bf16 (16B-aligned rows + bank skew)
constexpr int BS1_STR = kKS + 8;    // 136  (== TDM pad: 64 dwords + 4 dwords)
constexpr int HB_STR  = kHC + 8;    // 72
constexpr int WS2_STR = kHC + 8;    // 72   (== TDM pad: 32 dwords + 4 dwords)
constexpr int OFF_TOK  = 0;                                   // int[32]
constexpr int OFF_GATE = 128;                                 // float[32]
constexpr int OFF_XS   = 256;
constexpr int OFF_BS1  = OFF_XS + kMT * XS_STR * 2;           // 66304
constexpr int OFF_HB   = OFF_BS1 + kHC * BS1_STR * 2;         // 83712
constexpr int OFF_WS2  = OFF_HB + kMT * HB_STR * 2;           // 88320
constexpr int SMEM_BYTES = OFF_WS2 + 8 * 128 * WS2_STR * 2;   // 235776 (<320KB/WGP)

// ---------------- workspace layout (bytes) ----------------
constexpr size_t WOFF_CNT  = 0;
constexpr size_t WOFF_P    = 64;
constexpr size_t WOFF_IDX  = 128;
constexpr size_t WOFF_GL   = WOFF_IDX + (size_t)kE * kT * 4;          // 262272
constexpr size_t WOFF_XBF  = WOFF_GL + (size_t)kE * kT * 4;           // 524416
constexpr size_t WOFF_W1T  = WOFF_XBF + (size_t)kT * kD * 2;          // + 16 MB
constexpr size_t WOFF_W2T  = WOFF_W1T + (size_t)kE * kD * kH * 2;     // + 32 MB
constexpr size_t WOFF_SW1T = WOFF_W2T + (size_t)kE * kH * kD * 2;     // + 32 MB
constexpr size_t WOFF_SW2T = WOFF_SW1T + (size_t)kD * kH * 2;         // + 4 MB

// ---------------- WMMA types & helpers ----------------
typedef __attribute__((ext_vector_type(16))) __bf16    v16bf;
typedef __attribute__((ext_vector_type(8)))  float     v8f;
typedef __attribute__((ext_vector_type(4)))  int       v4i;
typedef __attribute__((ext_vector_type(8)))  int       v8i;
typedef __attribute__((ext_vector_type(4)))  unsigned  v4u;

struct U4x2 { uint4 lo, hi; };

__device__ __forceinline__ v16bf pack16(uint4 lo, uint4 hi) {
  U4x2 u; u.lo = lo; u.hi = hi;
  return __builtin_bit_cast(v16bf, u);
}

// A operand (16x32 bf16): lane<16 -> M=lane, K {0..7,16..23}; lane>=16 -> M=lane-16, K {8..15,24..31}
__device__ __forceinline__ v16bf load_a16(const __bf16* rowk, int lane) {
  const int off = (lane & 16) ? 8 : 0;
  uint4 lo = *reinterpret_cast<const uint4*>(rowk + off);
  uint4 hi = *reinterpret_cast<const uint4*>(rowk + off + 16);
  return pack16(lo, hi);
}

// B operand (32x16 bf16): lane<16 -> N=lane, K {0..7,16..23}; lane>=16 -> N=lane-16, K {8..15,24..31}
__device__ __forceinline__ v16bf load_b16(const __bf16* colk, int lane) {
  const int off = (lane & 16) ? 8 : 0;
  uint4 lo = *reinterpret_cast<const uint4*>(colk + off);
  uint4 hi = *reinterpret_cast<const uint4*>(colk + off + 16);
  return pack16(lo, hi);
}

__device__ __forceinline__ v8f wmma_bf16(v16bf a, v16bf b, v8f c) {
  return __builtin_amdgcn_wmma_f32_16x16x32_bf16(false, a, false, b, (short)0, c, false, false);
}

__device__ __forceinline__ __bf16 tobf(float f) { return (__bf16)f; }

// ---------------- async global->LDS 16B copy (CDNA5 ASYNCcnt path) ----------------
#if __has_builtin(__builtin_amdgcn_global_load_async_to_lds_b128)
#define ASYNC_LDS 1
#endif

typedef __attribute__((address_space(1))) v4i* gv4i_p;   // global int4*
typedef __attribute__((address_space(3))) v4i* lv4i_p;   // LDS int4*

__device__ __forceinline__ void async_cp16(const __bf16* g, __bf16* l) {
#ifdef ASYNC_LDS
  gv4i_p gp = (gv4i_p)(const_cast<__bf16*>(g));
  lv4i_p lp = (lv4i_p)(l);
  __builtin_amdgcn_global_load_async_to_lds_b128(gp, lp, 0, 0);
#else
  *reinterpret_cast<uint4*>(l) = *reinterpret_cast<const uint4*>(g);
#endif
}

__device__ __forceinline__ void async_wait0() {
#ifdef ASYNC_LDS
#if __has_builtin(__builtin_amdgcn_s_wait_asynccnt)
  __builtin_amdgcn_s_wait_asynccnt(0);
#else
  asm volatile("s_wait_asynccnt 0x0" ::: "memory");
#endif
#endif
}

// ---------------- TDM: whole-tile 2D global->LDS DMA (TENSORcnt path) ----------------
#if __has_builtin(__builtin_amdgcn_tensor_load_to_lds) && __has_builtin(__builtin_amdgcn_s_wait_tensorcnt)
#define TDM_LDS 1
#endif

#ifdef TDM_LDS
// 2D tile: tile_k contiguous elements (bf16) per row, tile_rows rows,
// row_stride_elems between rows in global; LDS rows padded by pad codes.
__device__ __forceinline__ void tdm_load_2d(const __bf16* gbase, unsigned lds_byte_addr,
                                            int tile_k, int tile_rows, int row_stride_elems,
                                            unsigned pad_interval_code, unsigned pad_amount_code) {
  unsigned long long ga = (unsigned long long)(size_t)gbase;
  v4u g0;
  g0[0] = 1u;                                                 // count=1, no gather
  g0[1] = lds_byte_addr;                                      // lds_addr
  g0[2] = (unsigned)(ga & 0xFFFFFFFFu);                       // global_addr[31:0]
  g0[3] = (unsigned)((ga >> 32) & 0x01FFFFFFu) | 0x80000000u; // global_addr[56:32] | type=2
  unsigned td0 = (unsigned)tile_k;      // tensor dims == tile dims (tiles never OOB)
  unsigned td1 = (unsigned)tile_rows;
  v8i g1;
  g1[0] = (int)((1u << 16)                         // data_size = 2 bytes
              | (1u << 20)                         // pad_enable
              | (pad_interval_code << 22)
              | (pad_amount_code << 25));
  g1[1] = (int)((td0 & 0xFFFFu) << 16);            // tensor_dim0[15:0] @63:48
  g1[2] = (int)((td0 >> 16) | ((td1 & 0xFFFFu) << 16));   // dim0[31:16] | dim1[15:0]
  g1[3] = (int)((td1 >> 16) | ((unsigned)tile_k << 16));  // dim1[31:16] | tile_dim0
  g1[4] = (int)(td1 & 0xFFFFu);                    // tile_dim1 (tile_dim2 = 0)
  g1[5] = (int)(unsigned)row_stride_elems;         // tensor_dim0_stride[31:0]
  g1[6] = 0;                                       // stride[47:32] = 0
  g1[7] = 0;
  v4i g2 = {0, 0, 0, 0};
  v4i g3 = {0, 0, 0, 0};
  v8i g4 = {0, 0, 0, 0, 0, 0, 0, 0};               // trailing group (6-arg toolchain form)
  __builtin_amdgcn_tensor_load_to_lds(g0, g1, g2, g3, g4, 0);
}
__device__ __forceinline__ unsigned lds_byte_addr_of(void* p) {
  return (unsigned)(size_t)((__attribute__((address_space(3))) char*)p);
}
#endif

// ---------------- kernel 0: zero counters ----------------
__global__ void moe_init_kernel(int* cnt, float* pmass) {
  int i = threadIdx.x;
  if (i < kE) { cnt[i] = 0; pmass[i] = 0.f; }
}

// ---------------- kernel 1: noisy gating + top-2 routing ----------------
__global__ __launch_bounds__(128) void moe_gate_kernel(
    const float* __restrict__ x, const float* __restrict__ noise,
    const float* __restrict__ wg, const float* __restrict__ wn,
    int* __restrict__ cnt, float* __restrict__ pmass,
    int* __restrict__ idx_list, float* __restrict__ gate_list) {
  const int t = blockIdx.x;
  const int tid = threadIdx.x;
  float pg[kE], pn[kE];
#pragma unroll
  for (int e = 0; e < kE; e++) { pg[e] = 0.f; pn[e] = 0.f; }
  const float* xr = x + (size_t)t * kD;
  for (int d = tid; d < kD; d += 128) {
    float xv = xr[d];
    const float* g = wg + d * kE;
    const float* n = wn + d * kE;
#pragma unroll
    for (int e = 0; e < kE; e++) { pg[e] += xv * g[e]; pn[e] += xv * n[e]; }
  }
  __shared__ float red[2 * kE][128];
#pragma unroll
  for (int e = 0; e < kE; e++) { red[e][tid] = pg[e]; red[kE + e][tid] = pn[e]; }
  __syncthreads();
  if (tid < 2 * kE) {
    float s = 0.f;
    for (int j = 0; j < 128; j++) s += red[tid][j];
    red[tid][0] = s;
  }
  __syncthreads();
  if (tid == 0) {
    float Hx[kE];
#pragma unroll
    for (int e = 0; e < kE; e++) {
      float gv = red[e][0];
      float nv = red[kE + e][0];
      float sp = (nv > 20.f) ? nv : log1pf(expf(nv));   // softplus
      Hx[e] = gv + noise[(size_t)t * kE + e] * sp;
    }
    int i0 = 0;
#pragma unroll
    for (int e = 1; e < kE; e++) if (Hx[e] > Hx[i0]) i0 = e;
    int i1 = -1;
#pragma unroll
    for (int e = 0; e < kE; e++) if (e != i0 && (i1 < 0 || Hx[e] > Hx[i1])) i1 = e;
    float m  = Hx[i0];
    float e1 = expf(Hx[i1] - m);
    float den = 1.f + e1;
    float g0 = 1.f / den, g1 = e1 / den;
    int p0 = atomicAdd(&cnt[i0], 1);
    idx_list[i0 * kT + p0]  = t;
    gate_list[i0 * kT + p0] = g0;
    atomicAdd(&pmass[i0], g0);
    int p1 = atomicAdd(&cnt[i1], 1);
    idx_list[i1 * kT + p1]  = t;
    gate_list[i1 * kT + p1] = g1;
    atomicAdd(&pmass[i1], g1);
  }
}

// ---------------- kernel 1b: fp32 -> bf16 elementwise (X) ----------------
struct alignas(8) bf16x4 { __bf16 a, b, c, d; };
__global__ __launch_bounds__(256) void moe_cvt_kernel(const float* __restrict__ src,
                                                      __bf16* __restrict__ dst) {
  int i = (blockIdx.x * 256 + threadIdx.x) * 4;
  float4 v = *reinterpret_cast<const float4*>(src + i);
  bf16x4 o{ (__bf16)v.x, (__bf16)v.y, (__bf16)v.z, (__bf16)v.w };
  *reinterpret_cast<bf16x4*>(dst + i) = o;
}

// ---------------- kernel 1c: tiled transpose + fp32->bf16 ----------------
// src: [z][rows][cols] fp32  ->  dst: [z][cols][rows] bf16
__global__ __launch_bounds__(256) void moe_transpose_cvt_kernel(
    const float* __restrict__ src, __bf16* __restrict__ dst, int rows, int cols) {
  __shared__ float tile[32][33];
  const size_t mat = (size_t)rows * cols;
  const float* s = src + blockIdx.z * mat;
  __bf16* d = dst + blockIdx.z * mat;
  const int bx = blockIdx.x * 32;   // col offset in src
  const int by = blockIdx.y * 32;   // row offset in src
  const int tx = threadIdx.x & 31;
  const int ty = threadIdx.x >> 5;  // 0..7
#pragma unroll
  for (int j = 0; j < 32; j += 8)
    tile[ty + j][tx] = s[(size_t)(by + ty + j) * cols + bx + tx];
  __syncthreads();
#pragma unroll
  for (int j = 0; j < 32; j += 8)
    d[(size_t)(bx + ty + j) * rows + by + tx] = (__bf16)tile[tx][ty + j];
}

// ---------------- kernel 2: fused expert FFN (fc1 -> ReLU -> fc2 -> gated combine) ----------------
// Weights pre-transposed bf16: W1T[e][h][d], W2T[e][d][h]; X pre-converted bf16 [t][d].
// Weight tiles staged by TDM (tensor_load_to_lds, TENSORcnt); X gather staged by
// per-lane async global->LDS B128 copies (ASYNCcnt).
__global__ __launch_bounds__(256) void moe_ffn_kernel(
    const __bf16* __restrict__ Xbf,
    const __bf16* __restrict__ W1T, const float* __restrict__ B1,
    const __bf16* __restrict__ W2T, const float* __restrict__ B2,
    const int* __restrict__ cnt, const int* __restrict__ idx_list,
    const float* __restrict__ gate_list,
    float* __restrict__ out, int shared_pass) {
  extern __shared__ char smem[];
  int*    tok_s   = reinterpret_cast<int*>(smem + OFF_TOK);
  float*  gates_s = reinterpret_cast<float*>(smem + OFF_GATE);
  __bf16* Xs      = reinterpret_cast<__bf16*>(smem + OFF_XS);    // [32][XS_STR]
  __bf16* Bs1     = reinterpret_cast<__bf16*>(smem + OFF_BS1);   // [64][BS1_STR]
  __bf16* Hb      = reinterpret_cast<__bf16*>(smem + OFF_HB);    // [32][HB_STR]
  __bf16* Ws2     = reinterpret_cast<__bf16*>(smem + OFF_WS2);   // [8][128][WS2_STR]

  const int tid  = threadIdx.x;
  const int wave = tid >> 5;
  const int lane = tid & 31;

  int e, row0, nrows;
  if (shared_pass) {
    e = 0; row0 = blockIdx.x * kMT; nrows = kMT;
  } else {
    e = blockIdx.x >> 8;                  // 256 tile slots per expert
    int tile = blockIdx.x & 255;
    int c = cnt[e];
    row0 = tile * kMT;
    if (row0 >= c) return;                // uniform per block
    nrows = min(kMT, c - row0);
  }
  const __bf16* W1Te = W1T + (size_t)e * kH * kD;   // [h][d]
  const __bf16* W2Te = W2T + (size_t)e * kD * kH;   // [d][h]
  const float*  B1e  = B1 + e * kH;
  const float*  B2e  = B2 + e * kD;

#ifdef TDM_LDS
  const unsigned ldsb_bs1 = lds_byte_addr_of(Bs1);
  const unsigned ldsb_ws2 = lds_byte_addr_of(Ws2);
#endif

  if (tid < kMT) {
    int t = -1; float g = 0.f;
    if (tid < nrows) {
      if (shared_pass) { t = row0 + tid; g = 1.f; }
      else { t = idx_list[e * kT + row0 + tid]; g = gate_list[e * kT + row0 + tid]; }
    }
    tok_s[tid] = t; gates_s[tid] = g;
  }
  __syncthreads();

  // stage X tile (token gather; async 16B copies; zero-fill rows past nrows)
  for (int i = tid; i < kMT * (kD / 8); i += 256) {
    int r = i >> 7, c8 = i & 127;
    int t = tok_s[r];
    __bf16* dst = Xs + r * XS_STR + c8 * 8;
    if (t >= 0) {
      async_cp16(Xbf + (size_t)t * kD + c8 * 8, dst);
    } else {
      uint4 z{0, 0, 0, 0};
      *reinterpret_cast<uint4*>(dst) = z;
    }
  }
  async_wait0();

  // fc1 tile ownership: 2 row-tiles x 4 col-tiles = 8 waves
  const int rt = wave & 1;
  const int ct = wave >> 1;

  v8f acc2[2][8];
#pragma unroll
  for (int a = 0; a < 2; a++)
#pragma unroll
    for (int b = 0; b < 8; b++) { v8f z = {}; acc2[a][b] = z; }

  for (int h0 = 0; h0 < kH; h0 += kHC) {
    // ---------- fc1: Hb[32][64] = relu(Xs @ W1[:, h0:h0+64] + b1) ----------
    v8f acc1 = {};
    for (int kb = 0; kb < kD; kb += kKS) {
      __syncthreads();                     // protect previous Bs1 readers (also covers Xs publish)
#ifdef TDM_LDS
      if (wave == 0) {
        // Bs1[c][k] = W1T[h0+c][kb+k]; 64 rows x 256B, pad 4 dwords / 64 dwords
        tdm_load_2d(W1Te + (size_t)h0 * kD + kb, ldsb_bs1, kKS, kHC, kD, 5u, 3u);
        if (kb == kD - kKS) {
          // Ws2[d][k] = W2T[d][h0+k]; 1024 rows x 128B, pad 4 dwords / 32 dwords.
          // Issued behind the last Bs1 tile: DMA overlaps the fc1 tail compute.
          tdm_load_2d(W2Te + h0, ldsb_ws2, kHC, kD, kH, 4u, 3u);
          __builtin_amdgcn_s_wait_tensorcnt(1);   // Bs1 done, Ws2 may stay in flight
        } else {
          __builtin_amdgcn_s_wait_tensorcnt(0);
        }
      }
#else
      for (int i = tid; i < kHC * (kKS / 8); i += 256) {
        int c = i >> 4, j = i & 15;
        async_cp16(W1Te + (size_t)(h0 + c) * kD + kb + j * 8, Bs1 + c * BS1_STR + j * 8);
      }
      async_wait0();
#endif
      __syncthreads();
#pragma unroll
      for (int kk = 0; kk < kKS; kk += 32) {
        v16bf a = load_a16(Xs + (rt * 16 + (lane & 15)) * XS_STR + kb + kk, lane);
        v16bf b = load_b16(Bs1 + (ct * 16 + (lane & 15)) * BS1_STR + kk, lane);
        acc1 = wmma_bf16(a, b, acc1);
      }
    }
    __syncthreads();                       // fc1 tiles done everywhere; Hb safe to rewrite
    {
      int n = ct * 16 + (lane & 15);
      float bias = B1e[h0 + n];
      int mbase = rt * 16 + ((lane & 16) ? 8 : 0);
#pragma unroll
      for (int i = 0; i < 8; i++) {
        float v = acc1[i] + bias;
        v = v > 0.f ? v : 0.f;
        Hb[(mbase + i) * HB_STR + n] = tobf(v);
      }
    }
    __bf16* Ws2w = Ws2 + wave * (128 * WS2_STR);
#ifdef TDM_LDS
    if (wave == 0) __builtin_amdgcn_s_wait_tensorcnt(0);   // Ws2 DMA complete
#else
    // stage this wave's private W2 slab: Ws2w[c][k] = W2T[wave*128+c][h0+k]
    for (int i = lane; i < 128 * (kHC / 8); i += 32) {
      int c = i >> 3, j = i & 7;
      async_cp16(W2Te + (size_t)(wave * 128 + c) * kH + h0 + j * 8, Ws2w + c * WS2_STR + j * 8);
    }
    async_wait0();
#endif
    __syncthreads();                       // publish Hb (and TDM Ws2) to all waves
    // ---------- fc2 accumulate: out[:, wave*128:+128] += Hb @ W2 chunk ----------
#pragma unroll
    for (int rt2 = 0; rt2 < 2; rt2++) {
#pragma unroll
      for (int kk = 0; kk < kHC; kk += 32) {
        v16bf a = load_a16(Hb + (rt2 * 16 + (lane & 15)) * HB_STR + kk, lane);
#pragma unroll
        for (int ct2 = 0; ct2 < 8; ct2++) {
          v16bf b = load_b16(Ws2w + (ct2 * 16 + (lane & 15)) * WS2_STR + kk, lane);
          acc2[rt2][ct2] = wmma_bf16(a, b, acc2[rt2][ct2]);
        }
      }
    }
  }

  // ---------- epilogue: gate-weighted combine ----------
  {
    const int nl = lane & 15;
    const int mo = (lane & 16) ? 8 : 0;
#pragma unroll
    for (int rt2 = 0; rt2 < 2; rt2++) {
#pragma unroll
      for (int ct2 = 0; ct2 < 8; ct2++) {
        int ncol = wave * 128 + ct2 * 16 + nl;
        float bias = B2e[ncol];
#pragma unroll
        for (int i = 0; i < 8; i++) {
          int m = rt2 * 16 + mo + i;
          if (m < nrows) {
            int t = tok_s[m];
            float val = acc2[rt2][ct2][i] + bias;
            if (shared_pass) out[(size_t)t * kD + ncol] = val;
            else atomicAdd(out + (size_t)t * kD + ncol, gates_s[m] * val);
          }
        }
      }
    }
  }
}

// ---------------- kernel 3: load-balance score ----------------
__global__ void moe_score_kernel(const int* __restrict__ cnt, const float* __restrict__ pmass,
                                 float* __restrict__ score) {
  if (threadIdx.x == 0) {
    float s = 0.f;
#pragma unroll
    for (int e = 0; e < kE; e++) s += (float)kE * (float)cnt[e] * pmass[e];
    float Tf = (float)kT;
    score[0] = s / (2.f * Tf * Tf) - 1.f;
  }
}

// ---------------- launcher ----------------
extern "C" void kernel_launch(void* const* d_in, const int* in_sizes, int n_in,
                              void* d_out, int out_size, void* d_ws, size_t ws_size,
                              hipStream_t stream) {
  const float* x     = (const float*)d_in[0];
  const float* noise = (const float*)d_in[1];
  const float* wg    = (const float*)d_in[2];
  const float* wn    = (const float*)d_in[3];
  const float* W1    = (const float*)d_in[4];
  const float* b1    = (const float*)d_in[5];
  const float* W2    = (const float*)d_in[6];
  const float* b2    = (const float*)d_in[7];
  const float* sW1   = (const float*)d_in[8];
  const float* sb1   = (const float*)d_in[9];
  const float* sW2   = (const float*)d_in[10];
  const float* sb2   = (const float*)d_in[11];
  float* out = (float*)d_out;

  char* ws = (char*)d_ws;
  int*    cnt       = (int*)(ws + WOFF_CNT);
  float*  pmass     = (float*)(ws + WOFF_P);
  int*    idx_list  = (int*)(ws + WOFF_IDX);
  float*  gate_list = (float*)(ws + WOFF_GL);
  __bf16* Xbf       = (__bf16*)(ws + WOFF_XBF);
  __bf16* W1Tb      = (__bf16*)(ws + WOFF_W1T);
  __bf16* W2Tb      = (__bf16*)(ws + WOFF_W2T);
  __bf16* sW1Tb     = (__bf16*)(ws + WOFF_SW1T);
  __bf16* sW2Tb     = (__bf16*)(ws + WOFF_SW2T);

  moe_init_kernel<<<1, 32, 0, stream>>>(cnt, pmass);
  moe_gate_kernel<<<kT, 128, 0, stream>>>(x, noise, wg, wn, cnt, pmass, idx_list, gate_list);

  // pre-pass: bf16 conversion (+ weight transposes so B staging is contiguous)
  moe_cvt_kernel<<<(kT * kD) / (256 * 4), 256, 0, stream>>>(x, Xbf);
  moe_transpose_cvt_kernel<<<dim3(kH / 32, kD / 32, kE), 256, 0, stream>>>(W1, W1Tb, kD, kH);
  moe_transpose_cvt_kernel<<<dim3(kD / 32, kH / 32, kE), 256, 0, stream>>>(W2, W2Tb, kH, kD);
  moe_transpose_cvt_kernel<<<dim3(kH / 32, kD / 32, 1), 256, 0, stream>>>(sW1, sW1Tb, kD, kH);
  moe_transpose_cvt_kernel<<<dim3(kD / 32, kH / 32, 1), 256, 0, stream>>>(sW2, sW2Tb, kH, kD);

  // shared expert: plain stores cover all T rows (acts as out initialization)
  moe_ffn_kernel<<<kT / kMT, 256, SMEM_BYTES, stream>>>(
      Xbf, sW1Tb, sb1, sW2Tb, sb2, nullptr, nullptr, nullptr, out, 1);
  // routed experts: gate-weighted atomic accumulation
  moe_ffn_kernel<<<kE * 256, 256, SMEM_BYTES, stream>>>(
      Xbf, W1Tb, b1, W2Tb, b2, cnt, idx_list, gate_list, out, 0);
  moe_score_kernel<<<1, 32, 0, stream>>>(cnt, pmass, out + (size_t)kT * kD);
}
